// SimpleRNN_5428838662592
// MI455X (gfx1250) — matrix operand order repaired and checked by reading, not compiled, not run on hardware
//
#include <hip/hip_runtime.h>
#include <hip/hip_bf16.h>
#include <math.h>

typedef __attribute__((ext_vector_type(16))) _Float16 v16h;
typedef __attribute__((ext_vector_type(8)))  float    v8f;

#define T_STEPS 32
#define BATCH   16
#define IN_DIM  112
#define HID     512
#define OUT_O   8
#define NGRP    7
#define E_SPLIT 409      // int(0.8*512)
#define EXIST_END 410    // H - round(0.2*H)
#define ALPHA_X 0.2f     // DT / 0.1
#define ALPHA_W 0.02f    // DT / 1.0
#define DT_C    0.02f
#define WB_C    10.0f

__device__ __forceinline__ float relu_f(float v) { return fmaxf(v, 0.f); }
__device__ __forceinline__ float clipw(float v)  { return fminf(fmaxf(v, 0.f), WB_C); }

__device__ __forceinline__ float wave_reduce(float s) {
#pragma unroll
  for (int off = 16; off > 0; off >>= 1) s += __shfl_down(s, off, 32);
  return s;
}

__global__ __launch_bounds__(256, 1)
void plastic_rnn_kernel(const float* __restrict__ x,      // (T,B,IN)
                        const float* __restrict__ R,      // (T,B)
                        const float* __restrict__ w_x2h,  // (H,IN)
                        const float* __restrict__ w_h2h,  // (H,H)
                        const float* __restrict__ b_h2h,  // (H)
                        const float* __restrict__ w_attn, // (NG,H)
                        const float* __restrict__ b_attn, // (NG)
                        const float* __restrict__ w_h2o,  // (O,H)
                        const float* __restrict__ w_h2v,  // (1,H)
                        const float* __restrict__ kappa,  // (21)
                        float* __restrict__ ys,           // (T,B,O+1)
                        float* __restrict__ ws)           // scratch: wh then wx per batch
{
  const int b    = blockIdx.x;
  const int tid  = threadIdx.x;
  const int wave = tid >> 5;
  const int lane = tid & 31;
  const int halfsel = lane >> 4;   // 0: K 0-7/16-23, 1: K 8-15/24-31 (A layout)
  const int mrow    = lane & 15;   // row within 16-row tile

  float* wh_b = ws + (size_t)b * HID * HID;
  float* wx_b = ws + (size_t)BATCH * HID * HID + (size_t)b * HID * IN_DIM;

  __shared__ float sh_wa[NGRP * HID];     // plastic wattn lives in LDS all kernel
  __shared__ float sh_out[2][HID];        // out(t) / out(t-1), parity-swapped
  __shared__ float sh_state[HID];
  __shared__ float sh_xg[2][IN_DIM];      // xg(t) / xg(t-1)
  __shared__ float sh_rec[HID];           // recurrent matvec accum
  __shared__ float sh_wxacc[HID];         // input matvec accum
  __shared__ float sh_diag[HID];          // wh[h,h] after update (for MASK_H diag)
  __shared__ float sh_alog[NGRP];
  __shared__ float sh_attn[NGRP];
  __shared__ float sh_kappa[24];
  __shared__ _Float16 __attribute__((aligned(32))) sh_so[HID]; // sign[j]*out(t)[j] as f16

  // ---- init ----
  for (int i = tid; i < NGRP * HID; i += 256) sh_wa[i] = relu_f(w_attn[i]);
  for (int j = tid; j < HID; j += 256) {
    sh_state[j] = 0.f; sh_out[0][j] = 0.f; sh_out[1][j] = 0.f; sh_diag[j] = 0.f;
  }
  if (tid < 24) sh_kappa[tid] = (tid < 21) ? kappa[tid] : 0.f;
  if (tid < IN_DIM) { sh_xg[0][tid] = 0.f; sh_xg[1][tid] = 0.f; }
  __syncthreads();

  for (int t = 0; t < T_STEPS; ++t) {
    const int p  = t & 1;        // out(t) buffer
    const int op = p ^ 1;        // out(t-1) buffer (becomes out(t+1) after phase D)
    const float dtR_old = (t > 0) ? DT_C * R[(t - 1) * BATCH + b] : 0.f;

    // ---- Phase A: attention logits (waves 0..6, one head each) ----
    if (wave < NGRP) {
      float s = 0.f;
      for (int j = lane; j < HID; j += 32) {
        float m = (j < E_SPLIT) ? 1.f : ((j < EXIST_END) ? -1.f : 0.f); // MASK_ATTN col
        s += sh_wa[wave * HID + j] * m * sh_out[p][j];
      }
      s = wave_reduce(s);
      if (lane == 0) sh_alog[wave] = s + b_attn[wave];
    }
    __syncthreads();
    if (tid == 0) {  // tiny softmax over 7
      float mx = sh_alog[0];
#pragma unroll
      for (int k = 1; k < NGRP; ++k) mx = fmaxf(mx, sh_alog[k]);
      float den = 0.f;
#pragma unroll
      for (int k = 0; k < NGRP; ++k) { float e = expf(sh_alog[k] - mx); sh_attn[k] = e; den += e; }
      float inv = 1.f / den;
#pragma unroll
      for (int k = 0; k < NGRP; ++k) sh_attn[k] *= inv;
    }
    __syncthreads();
    // gated input xg(t) and masked recurrent vector so(t)
    if (tid < IN_DIM)
      sh_xg[p][tid] = x[(size_t)(t * BATCH + b) * IN_DIM + tid] * sh_attn[tid >> 4];
    for (int j = tid; j < HID; j += 256) {
      float sgn = (j < E_SPLIT) ? 1.f : -1.f;
      sh_so[j] = (_Float16)(sgn * sh_out[p][j]);
    }
    __syncthreads();

    // ---- Phase B: fused wh update (pending from t-1) + WMMA matvec with out(t) ----
    // Each wave owns 4 row-tiles of 16 rows; per tile, 16 K-chunks of 32.
    for (int q = 0; q < 4; ++q) {
      const int rt  = wave * 4 + q;
      const int row = rt * 16 + mrow;
      const int rge = (row >= E_SPLIT) ? 1 : 0;
      const float u_row = sh_out[p][row];           // out_n(t-1)[row] = out(t)[row]
      const float* bhrow = w_h2h + (size_t)row * HID;
      float*       wsrow = wh_b + (size_t)row * HID;

      v8f acc = {};
#pragma unroll 1
      for (int kc = 0; kc < 16; ++kc) {
        const int c0 = kc * 32 + halfsel * 8;
        const int c1 = c0 + 16;
        if (kc < 15) __builtin_prefetch(bhrow + c0 + 32, 0, 3);

        float bb[16], wv[16];
        *(float4*)(bb + 0)  = *(const float4*)(bhrow + c0);
        *(float4*)(bb + 4)  = *(const float4*)(bhrow + c0 + 4);
        *(float4*)(bb + 8)  = *(const float4*)(bhrow + c1);
        *(float4*)(bb + 12) = *(const float4*)(bhrow + c1 + 4);

        if (t == 0) {
#pragma unroll
          for (int e = 0; e < 16; ++e) wv[e] = relu_f(bb[e]);   // W(0) = relu(base)
        } else {
          float ww[16];
          *(float4*)(ww + 0)  = *(const float4*)(wsrow + c0);
          *(float4*)(ww + 4)  = *(const float4*)(wsrow + c0 + 4);
          *(float4*)(ww + 8)  = *(const float4*)(wsrow + c1);
          *(float4*)(ww + 12) = *(const float4*)(wsrow + c1 + 4);
#pragma unroll
          for (int e = 0; e < 16; ++e) {
            const int col = (e < 8) ? (c0 + e) : (c1 + (e - 8));
            const float kr = sh_kappa[2 * rge + ((col >= E_SPLIT) ? 1 : 0)];
            const float term = dtR_old * kr * u_row * sh_out[op][col]; // out(t-1)[col]
            wv[e] = clipw(ww[e] - (ww[e] - relu_f(bb[e])) * ALPHA_W + term);
          }
        }
        // store updated W(t) tile back (one read + one write pass per step)
        *(float4*)(wsrow + c0)     = *(const float4*)(wv + 0);
        *(float4*)(wsrow + c0 + 4) = *(const float4*)(wv + 4);
        *(float4*)(wsrow + c1)     = *(const float4*)(wv + 8);
        *(float4*)(wsrow + c1 + 4) = *(const float4*)(wv + 12);

        // capture diagonal (MASK_H zeroes it) into LDS
#pragma unroll
        for (int e = 0; e < 16; ++e) {
          const int col = (e < 8) ? (c0 + e) : (c1 + (e - 8));
          if (col == row) sh_diag[row] = wv[e];
        }

        // A = updated tile (f16), B = so chunk broadcast across all 16 columns
        v16h av;
#pragma unroll
        for (int e = 0; e < 16; ++e) av[e] = (_Float16)wv[e];
        v16h bvv;
#pragma unroll
        for (int e = 0; e < 16; ++e) bvv[e] = sh_so[kc * 32 + halfsel * 8 + ((e < 8) ? e : (e + 8))];
        acc = __builtin_amdgcn_wmma_f32_16x16x32_f16(
            /*neg_a=*/false, av, /*neg_b=*/false, bvv,
            /*c_mod=*/(short)0, acc, /*reuse_a=*/false, /*reuse_b=*/false);
      }
      // every column of D equals the matvec; extract one copy per row
      if (lane == 0) {
#pragma unroll
        for (int r = 0; r < 8; ++r) sh_rec[rt * 16 + r] = acc[r];
      }
      if (lane == 16) {
#pragma unroll
        for (int r = 0; r < 8; ++r) sh_rec[rt * 16 + 8 + r] = acc[r];
      }
    }

    // ---- Phase C: fused wx update + matvec (VALU, float4 streams) ----
    for (int rr = 0; rr < 2; ++rr) {
      const int h   = tid + rr * 256;
      const int rge = (h >= E_SPLIT) ? 1 : 0;
      const float u_row = sh_out[p][h];
      float acc = 0.f;
      const float* bxrow = w_x2h + (size_t)h * IN_DIM;
      float*       wsxrow = wx_b + (size_t)h * IN_DIM;
#pragma unroll 4
      for (int i = 0; i < IN_DIM; i += 4) {
        const float4 bv = *(const float4*)(bxrow + i);
        float wv0, wv1, wv2, wv3;
        if (t == 0) {
          wv0 = relu_f(bv.x); wv1 = relu_f(bv.y); wv2 = relu_f(bv.z); wv3 = relu_f(bv.w);
        } else {
          const float4 ov = *(const float4*)(wsxrow + i);
          const float ki  = sh_kappa[4 + 2 * (i >> 4) + rge];
          const float fac = dtR_old * ki * u_row;
          wv0 = clipw(ov.x - (ov.x - relu_f(bv.x)) * ALPHA_W + fac * sh_xg[op][i + 0]);
          wv1 = clipw(ov.y - (ov.y - relu_f(bv.y)) * ALPHA_W + fac * sh_xg[op][i + 1]);
          wv2 = clipw(ov.z - (ov.z - relu_f(bv.z)) * ALPHA_W + fac * sh_xg[op][i + 2]);
          wv3 = clipw(ov.w - (ov.w - relu_f(bv.w)) * ALPHA_W + fac * sh_xg[op][i + 3]);
        }
        *(float4*)(wsxrow + i) = make_float4(wv0, wv1, wv2, wv3);
        acc += wv0 * sh_xg[p][i] + wv1 * sh_xg[p][i + 1] +
               wv2 * sh_xg[p][i + 2] + wv3 * sh_xg[p][i + 3];
      }
      sh_wxacc[h] = acc;
    }
    __syncthreads();

    // ---- Phase D: state/out update (subtract masked diagonal) ----
    for (int rr = 0; rr < 2; ++rr) {
      const int h = tid + rr * 256;
      const float sgn = (h < E_SPLIT) ? 1.f : -1.f;
      const float tot = sh_rec[h] - sh_diag[h] * sgn * sh_out[p][h]
                      + sh_wxacc[h] + b_h2h[h];
      const float st = (1.f - ALPHA_X) * sh_state[h] + ALPHA_X * tot;
      sh_state[h] = st;
      sh_out[op][h] = tanhf(fmaxf(st, 0.f));   // out(t+1)
    }
    __syncthreads();

    // ---- Phase E: wattn update in LDS (eager, uses attn(t), out(t+1), R(t)) ----
    {
      const float dtR = DT_C * R[t * BATCH + b];
      for (int idx = tid; idx < NGRP * HID; idx += 256) {
        const int k = idx >> 9;          // idx / 512
        const int h = idx & (HID - 1);
        const float basea = relu_f(w_attn[idx]);
        const float kfb = sh_kappa[18 + ((k < 3) ? 0 : ((k < 6) ? 1 : 2))];
        const float cm  = (h < E_SPLIT) ? 1.f : 0.f;
        const float wn = sh_wa[idx] - (sh_wa[idx] - basea) * ALPHA_W
                       + dtR * kfb * cm * sh_attn[k] * sh_out[op][h];
        sh_wa[idx] = clipw(wn);
      }
    }

    // ---- Phase F: outputs o_t (waves 0..7) and v_t (wave 0) ----
    if (wave < OUT_O) {
      float s = 0.f;
      for (int j = lane; j < HID; j += 32) {
        const float ex = (j < EXIST_END) ? 1.f : 0.f;
        s += relu_f(w_h2o[wave * HID + j]) * ex * sh_out[op][j];
      }
      s = wave_reduce(s);
      if (lane == 0) ys[(size_t)(t * BATCH + b) * (OUT_O + 1) + wave] = s;
    }
    if (wave == 0) {
      float s = 0.f;
      for (int j = lane; j < HID; j += 32) {
        const float ex = (j < EXIST_END) ? 1.f : 0.f;
        s += relu_f(w_h2v[j]) * ex * sh_out[op][j];
      }
      s = wave_reduce(s);
      if (lane == 0) ys[(size_t)(t * BATCH + b) * (OUT_O + 1) + OUT_O] = s;
    }
    __syncthreads();
  }
}

extern "C" void kernel_launch(void* const* d_in, const int* in_sizes, int n_in,
                              void* d_out, int out_size, void* d_ws, size_t ws_size,
                              hipStream_t stream) {
  (void)in_sizes; (void)n_in; (void)out_size; (void)ws_size;
  const float* x      = (const float*)d_in[0];
  const float* R      = (const float*)d_in[1];
  const float* w_x2h  = (const float*)d_in[2];
  const float* w_h2h  = (const float*)d_in[3];
  const float* b_h2h  = (const float*)d_in[4];
  const float* w_attn = (const float*)d_in[5];
  const float* b_attn = (const float*)d_in[6];
  const float* w_h2o  = (const float*)d_in[7];
  const float* w_h2v  = (const float*)d_in[8];
  const float* kappa  = (const float*)d_in[9];

  plastic_rnn_kernel<<<dim3(BATCH), dim3(256), 0, stream>>>(
      x, R, w_x2h, w_h2h, b_h2h, w_attn, b_attn, w_h2o, w_h2v, kappa,
      (float*)d_out, (float*)d_ws);
}
